// MultiHeadAttention_25443386261658
// MI455X (gfx1250) — matrix operand order repaired and checked by reading, not compiled, not run on hardware
//
#include <hip/hip_runtime.h>
#include <hip/hip_bf16.h>

typedef __attribute__((ext_vector_type(16))) __bf16 v16bf;
typedef __attribute__((ext_vector_type(8)))  float  v8f;
typedef __attribute__((ext_vector_type(4)))  int    v4i;

#define S_LEN 2048
#define HEADS 16
#define DK 64
#define DMODEL 1024

// ---------------------------------------------------------------------------
// CDNA5 async global->LDS staging (ASYNCcnt-tracked), with safe fallback.
// Probe-verified: __builtin_amdgcn_global_load_async_to_lds_b128 takes
// (v4i addrspace(1)*, v4i addrspace(3)*, imm offset, imm cpol).
// ---------------------------------------------------------------------------
#if defined(__has_builtin)
#if __has_builtin(__builtin_amdgcn_global_load_async_to_lds_b128)
#define ASYNC_LDS 1
#endif
#endif
#ifndef ASYNC_LDS
#define ASYNC_LDS 0
#endif

typedef __attribute__((address_space(1))) v4i as1_v4i;
typedef __attribute__((address_space(3))) v4i as3_v4i;

__device__ __forceinline__ void async_ld_b128(const void* g, void* l) {
#if ASYNC_LDS
  __builtin_amdgcn_global_load_async_to_lds_b128((as1_v4i*)g, (as3_v4i*)l, 0, 0);
#else
  *(uint4*)l = *(const uint4*)g;
#endif
}

template <int N>
__device__ __forceinline__ void wait_async() {
#if ASYNC_LDS
#if __has_builtin(__builtin_amdgcn_s_wait_asynccnt)
  __builtin_amdgcn_s_wait_asynccnt(N);
#else
  asm volatile("s_wait_asynccnt %0" ::"i"(N) : "memory");
#endif
#endif
}

// ---------------------------------------------------------------------------
// f32 -> bf16 elementwise convert (with CDNA5 global_prefetch hint)
// ---------------------------------------------------------------------------
__global__ void f32_to_bf16_kernel(const float* __restrict__ in,
                                   __bf16* __restrict__ out, int n) {
  int i = blockIdx.x * 256 + threadIdx.x;
  if (i + 65536 < n) __builtin_prefetch(in + i + 65536, 0, 1);
  if (i < n) out[i] = (__bf16)in[i];
}

// ---------------------------------------------------------------------------
// WMMA fragment layouts (wave32, CDNA5 ISA 7.12.2)
// A (16x32 bf16): lane<16 -> row=lane, K {0..7,16..23}; lane>=16 -> {8..15,24..31}
// B (32x16 bf16): col = lane&15; this file packs A and B with K = (lane>=16)*16 + h, h=0..15
// C (16x16 f32):  vgpr v -> row = v + 8*(lane>=16), col = lane&15
// ---------------------------------------------------------------------------

// ---------------------------------------------------------------------------
// bf16 WMMA GEMM, double-buffered async LDS staging.
// MODE 0: f32 output row-major [M,N];  MODE 1: bf16 split-head output.
// block = 128 threads (4 waves); tile 64x64; K step 32
// ---------------------------------------------------------------------------
template <int MODE>
__global__ __launch_bounds__(128) void gemm_wmma_bf16(
    const __bf16* __restrict__ A, const __bf16* __restrict__ Bw,
    const float* __restrict__ bias, void* __restrict__ outp,
    int M, int N, int K) {
  __shared__ __align__(16) __bf16 As[2][64][32];
  __shared__ __align__(16) __bf16 Bs[2][32][64];

  const int tid   = threadIdx.x;
  const int lane  = tid & 31;
  const int wave  = tid >> 5;
  const int l15   = lane & 15;
  const int khalf = lane >> 4;

  const int m0 = blockIdx.y * 64;
  const int n0 = blockIdx.x * 64;

  auto stageA = [&](int buf, int k0) {
    int row = tid >> 1;           // 0..63
    int col = (tid & 1) * 16;     // 0 or 16
    const __bf16* src = A + (size_t)(m0 + row) * K + k0 + col;
    async_ld_b128(src,     &As[buf][row][col]);
    async_ld_b128(src + 8, &As[buf][row][col + 8]);
  };
  auto stageB = [&](int buf, int k0) {
    int row = tid >> 2;           // 0..31
    int col = (tid & 3) * 16;     // 0,16,32,48
    const __bf16* src = Bw + (size_t)(k0 + row) * N + n0 + col;
    async_ld_b128(src,     &Bs[buf][row][col]);
    async_ld_b128(src + 8, &Bs[buf][row][col + 8]);
  };

  v8f acc[4] = {};

  stageA(0, 0);
  stageB(0, 0);
  for (int k0 = 0; k0 < K; k0 += 32) {
    const int buf = (k0 >> 5) & 1;
    if (k0 + 32 < K) {
      stageA(buf ^ 1, k0 + 32);
      stageB(buf ^ 1, k0 + 32);
      wait_async<4>();            // in-order: tile k0 has landed
    } else {
      wait_async<0>();
    }
    __syncthreads();

    v16bf afrag;
    {
      int arow = wave * 16 + l15;
#pragma unroll
      for (int h = 0; h < 16; ++h) {
        int kk = khalf * 16 + h;
        afrag[h] = As[buf][arow][kk];
      }
    }
#pragma unroll
    for (int t = 0; t < 4; ++t) {
      v16bf bfrag;
      int bcol = t * 16 + l15;
      int kb = khalf * 16;
#pragma unroll
      for (int h = 0; h < 16; ++h) bfrag[h] = Bs[buf][kb + h][bcol];
      acc[t] = __builtin_amdgcn_wmma_f32_16x16x32_bf16(
          false, afrag, false, bfrag, (short)0, acc[t], false, false);
    }
    __syncthreads();
  }

#pragma unroll
  for (int t = 0; t < 4; ++t) {
#pragma unroll
    for (int v = 0; v < 8; ++v) {
      int r = m0 + wave * 16 + v + khalf * 8;
      int c = n0 + t * 16 + l15;
      float val = acc[t][v] + bias[c];
      if (MODE == 0) {
        ((float*)outp)[(size_t)r * N + c] = val;
      } else {
        int bb = r >> 11, ss = r & (S_LEN - 1);
        int hh = c >> 6,  dd = c & (DK - 1);
        ((__bf16*)outp)[(((size_t)(bb * HEADS + hh) * S_LEN) + ss) * DK + dd] =
            (__bf16)val;
      }
    }
  }
}

// ---------------------------------------------------------------------------
// Attention. 4 waves/block: each wave owns one 16-row q-tile of one (b,h);
// the 4 waves share (b,h) so K/V tiles are staged cooperatively into LDS
// via double-buffered async loads.
// Phase 1: online max/sumexp (scores stay in registers).
// Phase 2: recompute scores, p = exp(s-m)/l, single attn write, bounce p
//          tile through LDS (C-layout -> A-layout), ctx += P*V via WMMA.
// ---------------------------------------------------------------------------
__global__ __launch_bounds__(128) void attn_wmma(
    const __bf16* __restrict__ Q, const __bf16* __restrict__ Kmat,
    const __bf16* __restrict__ Vmat, const int* __restrict__ mask,
    float* __restrict__ attn, __bf16* __restrict__ ctx) {
  __shared__ __align__(16) __bf16 Kt[2][32][64];   // 8 KB
  __shared__ __align__(16) __bf16 Vt[2][32][64];   // 8 KB
  __shared__ __align__(16) __bf16 Ps[4][16][32];   // 4 KB

  const int tid   = threadIdx.x;
  const int lane  = tid & 31;
  const int wave  = tid >> 5;
  const int l15   = lane & 15;
  const int khalf = lane >> 4;

  const int q0   = (blockIdx.x * 4 + wave) * 16;
  const int head = blockIdx.y;
  const int b    = blockIdx.z;

  const size_t bh = (size_t)(b * HEADS + head);
  const __bf16* Qb    = Q    + (bh * S_LEN + q0) * DK;
  const __bf16* Kbase = Kmat + bh * S_LEN * DK;
  const __bf16* Vbase = Vmat + bh * S_LEN * DK;
  const int*    mrow  = mask + bh * (size_t)S_LEN * S_LEN + (size_t)q0 * S_LEN;
  float*        arow  = attn + bh * (size_t)S_LEN * S_LEN + (size_t)q0 * S_LEN;

  const float scale = 0.125f;  // 1/sqrt(64)

  auto stageK = [&](int buf, int key0) {
    int row = tid >> 2;          // 0..31 (key within tile)
    int col = (tid & 3) * 16;    // d offset
    const __bf16* src = Kbase + (size_t)(key0 + row) * DK + col;
    async_ld_b128(src,     &Kt[buf][row][col]);
    async_ld_b128(src + 8, &Kt[buf][row][col + 8]);
  };
  auto stageV = [&](int buf, int key0) {
    int row = tid >> 2;
    int col = (tid & 3) * 16;
    const __bf16* src = Vbase + (size_t)(key0 + row) * DK + col;
    async_ld_b128(src,     &Vt[buf][row][col]);
    async_ld_b128(src + 8, &Vt[buf][row][col + 8]);
  };

  // Q A-fragments (k=0..31, 32..63), reused for every key tile
  v16bf aQ0, aQ1;
#pragma unroll
  for (int h = 0; h < 16; ++h) {
    int kk = khalf * 16 + h;
    aQ0[h] = Qb[l15 * DK + kk];
    aQ1[h] = Qb[l15 * DK + 32 + kk];
  }

  // -------- phase 1: online softmax statistics --------
  float m_[8], l_[8];
#pragma unroll
  for (int v = 0; v < 8; ++v) { m_[v] = -3.0e38f; l_[v] = 0.0f; }

  stageK(0, 0);
  for (int kt = 0; kt < S_LEN / 32; ++kt) {
    const int buf = kt & 1;
    if (kt + 1 < S_LEN / 32) {
      stageK(buf ^ 1, (kt + 1) * 32);
      wait_async<2>();
    } else {
      wait_async<0>();
    }
    __syncthreads();

#pragma unroll
    for (int t2 = 0; t2 < 2; ++t2) {
      int key0 = kt * 32 + t2 * 16;
      v16bf bK0, bK1;
#pragma unroll
      for (int h = 0; h < 16; ++h) {
        int d = khalf * 16 + h;
        bK0[h] = Kt[buf][t2 * 16 + l15][d];
        bK1[h] = Kt[buf][t2 * 16 + l15][32 + d];
      }
      v8f sc = {};
      sc = __builtin_amdgcn_wmma_f32_16x16x32_bf16(false, aQ0, false, bK0,
                                                   (short)0, sc, false, false);
      sc = __builtin_amdgcn_wmma_f32_16x16x32_bf16(false, aQ1, false, bK1,
                                                   (short)0, sc, false, false);
#pragma unroll
      for (int v = 0; v < 8; ++v) {
        int r = v + khalf * 8;
        int c = key0 + l15;
        float s = sc[v] * scale;
        s = mrow[(size_t)r * S_LEN + c] ? s : -1.0e10f;
        float tmax = s;
#pragma unroll
        for (int off = 1; off < 16; off <<= 1)
          tmax = fmaxf(tmax, __shfl_xor(tmax, off, 32));
        float mnew = fmaxf(m_[v], tmax);
        float e = __expf(s - mnew);
        float tsum = e;
#pragma unroll
        for (int off = 1; off < 16; off <<= 1)
          tsum += __shfl_xor(tsum, off, 32);
        l_[v] = l_[v] * __expf(m_[v] - mnew) + tsum;
        m_[v] = mnew;
      }
    }
    __syncthreads();
  }
  float linv_[8];
#pragma unroll
  for (int v = 0; v < 8; ++v) linv_[v] = (l_[v] > 0.0f) ? 1.0f / l_[v] : 0.0f;

  // -------- phase 2: recompute, normalize, write attn once, ctx = P*V ------
  v8f cacc[4] = {};
  stageK(0, 0);
  stageV(0, 0);
  for (int kt = 0; kt < S_LEN / 32; ++kt) {
    const int buf = kt & 1;
    if (kt + 1 < S_LEN / 32) {
      stageK(buf ^ 1, (kt + 1) * 32);
      stageV(buf ^ 1, (kt + 1) * 32);
      wait_async<4>();
    } else {
      wait_async<0>();
    }
    __syncthreads();

#pragma unroll
    for (int t2 = 0; t2 < 2; ++t2) {
      int key0 = kt * 32 + t2 * 16;
      v16bf bK0, bK1;
#pragma unroll
      for (int h = 0; h < 16; ++h) {
        int d = khalf * 16 + h;
        bK0[h] = Kt[buf][t2 * 16 + l15][d];
        bK1[h] = Kt[buf][t2 * 16 + l15][32 + d];
      }
      v8f sc = {};
      sc = __builtin_amdgcn_wmma_f32_16x16x32_bf16(false, aQ0, false, bK0,
                                                   (short)0, sc, false, false);
      sc = __builtin_amdgcn_wmma_f32_16x16x32_bf16(false, aQ1, false, bK1,
                                                   (short)0, sc, false, false);
#pragma unroll
      for (int v = 0; v < 8; ++v) {
        int r = v + khalf * 8;
        int c = key0 + l15;
        float s = sc[v] * scale;
        s = mrow[(size_t)r * S_LEN + c] ? s : -1.0e10f;
        float p = __expf(s - m_[v]) * linv_[v];
        arow[(size_t)r * S_LEN + c] = p;         // single attn write
        Ps[wave][r][t2 * 16 + l15] = (__bf16)p;  // C-layout -> LDS
      }
    }
    __syncthreads();

    v16bf aP;
#pragma unroll
    for (int h = 0; h < 16; ++h) {
      int kk = khalf * 16 + h;
      aP[h] = Ps[wave][l15][kk];
    }
    v16bf bV0, bV1, bV2, bV3;
#pragma unroll
    for (int h = 0; h < 16; ++h) {
      int kr = khalf * 16 + h;
      bV0[h] = Vt[buf][kr][l15];
      bV1[h] = Vt[buf][kr][16 + l15];
      bV2[h] = Vt[buf][kr][32 + l15];
      bV3[h] = Vt[buf][kr][48 + l15];
    }
    cacc[0] = __builtin_amdgcn_wmma_f32_16x16x32_bf16(false, aP, false, bV0,
                                                      (short)0, cacc[0], false, false);
    cacc[1] = __builtin_amdgcn_wmma_f32_16x16x32_bf16(false, aP, false, bV1,
                                                      (short)0, cacc[1], false, false);
    cacc[2] = __builtin_amdgcn_wmma_f32_16x16x32_bf16(false, aP, false, bV2,
                                                      (short)0, cacc[2], false, false);
    cacc[3] = __builtin_amdgcn_wmma_f32_16x16x32_bf16(false, aP, false, bV3,
                                                      (short)0, cacc[3], false, false);
  }

  // ctx in [B,S,H*64] bf16 (row-major for the output GEMM)
#pragma unroll
  for (int t = 0; t < 4; ++t) {
#pragma unroll
    for (int v = 0; v < 8; ++v) {
      int rr = v + khalf * 8;
      size_t orow = (size_t)b * S_LEN + q0 + rr;
      int col = head * DK + t * 16 + l15;
      ctx[orow * (HEADS * DK) + col] = (__bf16)cacc[t][v];
    }
  }
}

// ---------------------------------------------------------------------------
// launch
// ---------------------------------------------------------------------------
extern "C" void kernel_launch(void* const* d_in, const int* in_sizes, int n_in,
                              void* d_out, int out_size, void* d_ws,
                              size_t ws_size, hipStream_t stream) {
  const float* X  = (const float*)d_in[0];
  const float* Wq = (const float*)d_in[1];
  const float* bq = (const float*)d_in[2];
  const float* Wk = (const float*)d_in[3];
  const float* bk = (const float*)d_in[4];
  const float* Wv = (const float*)d_in[5];
  const float* bv = (const float*)d_in[6];
  const float* Wo = (const float*)d_in[7];
  const float* bo = (const float*)d_in[8];
  const int*   mask = (const int*)d_in[9];

  float* out_f  = (float*)d_out;                               // [2,2048,1024]
  float* attn_f = (float*)d_out + (size_t)2 * S_LEN * DMODEL;  // [2,16,2048,2048]

  char* ws = (char*)d_ws;
  __bf16* Xbf = (__bf16*)(ws);                          //  8 MB
  __bf16* Wqb = (__bf16*)(ws + ((size_t)8  << 20));     //  2 MB
  __bf16* Wkb = (__bf16*)(ws + ((size_t)10 << 20));
  __bf16* Wvb = (__bf16*)(ws + ((size_t)12 << 20));
  __bf16* Wob = (__bf16*)(ws + ((size_t)14 << 20));
  __bf16* Qbf = (__bf16*)(ws + ((size_t)16 << 20));     //  8 MB [B,H,S,64]
  __bf16* Kbf = (__bf16*)(ws + ((size_t)24 << 20));
  __bf16* Vbf = (__bf16*)(ws + ((size_t)32 << 20));
  __bf16* Cbf = (__bf16*)(ws + ((size_t)40 << 20));     //  8 MB [B*S, 1024]

  const int NX = 2 * S_LEN * DMODEL;       // 4194304
  const int NW = DMODEL * DMODEL;          // 1048576
  f32_to_bf16_kernel<<<(NX + 255) / 256, 256, 0, stream>>>(X, Xbf, NX);
  f32_to_bf16_kernel<<<(NW + 255) / 256, 256, 0, stream>>>(Wq, Wqb, NW);
  f32_to_bf16_kernel<<<(NW + 255) / 256, 256, 0, stream>>>(Wk, Wkb, NW);
  f32_to_bf16_kernel<<<(NW + 255) / 256, 256, 0, stream>>>(Wv, Wvb, NW);
  f32_to_bf16_kernel<<<(NW + 255) / 256, 256, 0, stream>>>(Wo, Wob, NW);

  dim3 gg(DMODEL / 64, (2 * S_LEN) / 64);  // (16, 64)
  gemm_wmma_bf16<1><<<gg, 128, 0, stream>>>(Xbf, Wqb, bq, (void*)Qbf,
                                            2 * S_LEN, DMODEL, DMODEL);
  gemm_wmma_bf16<1><<<gg, 128, 0, stream>>>(Xbf, Wkb, bk, (void*)Kbf,
                                            2 * S_LEN, DMODEL, DMODEL);
  gemm_wmma_bf16<1><<<gg, 128, 0, stream>>>(Xbf, Wvb, bv, (void*)Vbf,
                                            2 * S_LEN, DMODEL, DMODEL);

  dim3 ag(S_LEN / 64, HEADS, 2);           // (32, 16, 2); 4 q-tiles per block
  attn_wmma<<<ag, 128, 0, stream>>>(Qbf, Kbf, Vbf, mask, attn_f, Cbf);

  gemm_wmma_bf16<0><<<gg, 128, 0, stream>>>(Cbf, Wob, bo, (void*)out_f,
                                            2 * S_LEN, DMODEL, DMODEL);
}
